// DecoderVAE_15453292331078
// MI455X (gfx1250) — matrix operand order repaired and checked by reading, not compiled, not run on hardware
//
#include <hip/hip_runtime.h>
#include <hip/hip_bf16.h>
#include <math.h>

// ---------------- problem constants ----------------
#define Hdim 512
#define Zdim 128
#define Edim 512
#define Ddim 1152          // 2*H + Z
#define Vdim 32000
#define BS   256
#define Ndim 128

typedef __bf16 bf16_t;
typedef __attribute__((ext_vector_type(16))) __bf16 v16bf;
typedef __attribute__((ext_vector_type(8)))  float  v8f;

// ---------------- WMMA fragment loaders (ISA 7.12.2 layouts, wave32) -------
// A (16-bit, 16x32 MxK): lane l -> row (l&15); K = e + (e<8?0:8) + 8*(l>=16)
__device__ __forceinline__ v16bf load_a_frag(const float* __restrict__ A, int lda,
                                             int rowBase, int k0, int lane) {
    const float* row = A + (size_t)(rowBase + (lane & 15)) * lda + k0 + ((lane >> 4) * 8);
    v16bf a;
#pragma unroll
    for (int e = 0; e < 8; ++e)  a[e]     = (bf16_t)row[e];
#pragma unroll
    for (int e = 0; e < 8; ++e)  a[8 + e] = (bf16_t)row[16 + e];
    return a;
}

// B (16-bit, 32x16 KxN) for C = A * B^T with row-major Bmat(N,K):
// lane l -> col (l&15); K = e + (e<8?0:8) + 8*(l>=16)  (same as A).
__device__ __forceinline__ v16bf load_b_frag(const float* __restrict__ B, int ldb,
                                             int nBase, int k0, int lane) {
    const float* row = B + (size_t)(nBase + (lane & 15)) * ldb + k0 + ((lane >> 4) * 8);
    v16bf b;
#pragma unroll
    for (int e = 0; e < 8; ++e)  b[e]     = (bf16_t)row[e];
#pragma unroll
    for (int e = 0; e < 8; ++e)  b[8 + e] = (bf16_t)row[16 + e];
    return b;
}

__device__ __forceinline__ float sigmoidf(float x) { return 1.0f / (1.0f + __expf(-x)); }

// ---------------- generic C = A(M,K) * B(Nn,K)^T + bias --------------------
// block = 128 threads (4 waves); wave w owns 16x16 tile; grid = (M/16, Nn/64)
__global__ void gemm_bt_bias(const float* __restrict__ A, const float* __restrict__ B,
                             const float* __restrict__ bias, float* __restrict__ C,
                             int M, int Nn, int K) {
    const int lane  = threadIdx.x & 31;
    const int wave  = threadIdx.x >> 5;
    const int rowB  = blockIdx.x * 16;
    const int nB    = blockIdx.y * 64 + wave * 16;

    v8f acc = {0.f, 0.f, 0.f, 0.f, 0.f, 0.f, 0.f, 0.f};
    for (int k0 = 0; k0 < K; k0 += 32) {
        v16bf a = load_a_frag(A, K, rowB, k0, lane);
        v16bf b = load_b_frag(B, K, nB, k0, lane);
        acc = __builtin_amdgcn_wmma_f32_16x16x32_bf16(false, a, false, b,
                                                      (short)0, acc, false, false);
    }
    const int hi  = lane >> 4;
    const int col = nB + (lane & 15);
    const float bv = bias ? bias[col] : 0.0f;
#pragma unroll
    for (int r = 0; r < 8; ++r) {
        const int row = rowB + r + 8 * hi;
        C[(size_t)row * Nn + col] = acc[r] + bv;
    }
}

// ---------- fused attention scores: u[b,n] = sum_k tanh(et+dt)*vt ----------
// et tile computed by WMMA; enc (huge operand) streamed exactly once:
// A fragment loaded once per K-step, reused across 8 column tiles per wave.
// grid = BS*N/16 row tiles, block = 128 (4 waves x 8 col tiles = 512 cols).
__global__ void attn_scores(const float* __restrict__ enc,   // (BS*N, H)
                            const float* __restrict__ W1,    // (H, H)
                            const float* __restrict__ dt,    // (BS, H)
                            const float* __restrict__ vt,    // (H)
                            float* __restrict__ u)           // (BS*N)
{
    const int lane = threadIdx.x & 31;
    const int wave = threadIdx.x >> 5;
    const int rowB = blockIdx.x * 16;          // rows share one batch b (N%16==0)
    const int b    = rowB / Ndim;

    __shared__ float su[16];
    if (threadIdx.x < 16) su[threadIdx.x] = 0.0f;
    __syncthreads();

    v8f acc[8] = {};
    for (int k0 = 0; k0 < Hdim; k0 += 32) {
        v16bf a = load_a_frag(enc, Hdim, rowB, k0, lane);
#pragma unroll
        for (int t = 0; t < 8; ++t) {
            const int nB = (t * 4 + wave) * 16;
            v16bf bf = load_b_frag(W1, Hdim, nB, k0, lane);
            acc[t] = __builtin_amdgcn_wmma_f32_16x16x32_bf16(false, a, false, bf,
                                                             (short)0, acc[t], false, false);
        }
    }
    const int hi = lane >> 4;
    float rowsum[8];
#pragma unroll
    for (int r = 0; r < 8; ++r) rowsum[r] = 0.0f;
#pragma unroll
    for (int t = 0; t < 8; ++t) {
        const int col = (t * 4 + wave) * 16 + (lane & 15);
        const float d = dt[(size_t)b * Hdim + col];
        const float v = vt[col];
#pragma unroll
        for (int r = 0; r < 8; ++r)
            rowsum[r] += tanhf(acc[t][r] + d) * v;
    }
#pragma unroll
    for (int r = 0; r < 8; ++r)
        atomicAdd(&su[r + 8 * hi], rowsum[r]);
    __syncthreads();
    if (threadIdx.x < 16)
        u[(size_t)rowB + threadIdx.x] = su[threadIdx.x];
}

// ------------- masked softmax over N=128 (mask true -> -inf) ---------------
__global__ void masked_softmax(const float* __restrict__ u, const int* __restrict__ mask,
                               float* __restrict__ aw) {
    const int b = blockIdx.x, t = threadIdx.x;       // blockDim = 128
    float val = mask[b * Ndim + t] ? -__builtin_inff() : u[b * Ndim + t];
    __shared__ float red[128];
    red[t] = val; __syncthreads();
    for (int s = 64; s > 0; s >>= 1) { if (t < s) red[t] = fmaxf(red[t], red[t + s]); __syncthreads(); }
    const float m = red[0]; __syncthreads();
    const float e = __expf(val - m);
    red[t] = e; __syncthreads();
    for (int s = 64; s > 0; s >>= 1) { if (t < s) red[t] += red[t + s]; __syncthreads(); }
    aw[b * Ndim + t] = e / red[0];
}

// ----------------- ctx[b,h] = sum_n aw[b,n] * enc[b,n,h] -------------------
__global__ void weighted_ctx(const float* __restrict__ aw, const float* __restrict__ enc,
                             float* __restrict__ ctx) {
    const int b = blockIdx.x;
    const int h = blockIdx.y * blockDim.x + threadIdx.x;   // grid.y=2, block=256
    float s = 0.0f;
    for (int n = 0; n < Ndim; ++n)
        s += aw[b * Ndim + n] * enc[((size_t)b * Ndim + n) * Hdim + h];
    ctx[(size_t)b * Hdim + h] = s;
}

// -------- plan = softmax(prev_h @ plan_W^T + plan_b)  (2 logits) -----------
__global__ void plan_kernel(const float* __restrict__ prev_h, const float* __restrict__ plan_W,
                            const float* __restrict__ plan_b, float* __restrict__ plan,
                            float* __restrict__ plan_out) {
    const int b = blockIdx.x, t = threadIdx.x;             // blockDim = 128
    float s0 = 0.0f, s1 = 0.0f;
    for (int d = t; d < Ddim; d += 128) {
        const float hv = prev_h[(size_t)b * Ddim + d];
        s0 += hv * plan_W[d];
        s1 += hv * plan_W[Ddim + d];
    }
    __shared__ float r0[128], r1[128];
    r0[t] = s0; r1[t] = s1; __syncthreads();
    for (int s = 64; s > 0; s >>= 1) {
        if (t < s) { r0[t] += r0[t + s]; r1[t] += r1[t + s]; }
        __syncthreads();
    }
    if (t == 0) {
        const float a = r0[0] + plan_b[0], c = r1[0] + plan_b[1];
        const float m = fmaxf(a, c);
        const float ea = __expf(a - m), ec = __expf(c - m);
        const float inv = 1.0f / (ea + ec);
        plan[b * 2 + 0] = ea * inv;  plan[b * 2 + 1] = ec * inv;
        plan_out[b * 2 + 0] = ea * inv;  plan_out[b * 2 + 1] = ec * inv;
    }
}

// ------- y_ctx = [prev_y, plan0*ctx_e + plan1*ctx_s]  (BS, E+H) ------------
__global__ void build_yctx(const float* __restrict__ prev_y, const float* __restrict__ plan,
                           const float* __restrict__ ctx_e, const float* __restrict__ ctx_s,
                           float* __restrict__ y_ctx) {
    const int b = blockIdx.x;
    const int i = blockIdx.y * blockDim.x + threadIdx.x;   // grid.y=4, block=256 -> 1024
    float v;
    if (i < Edim) {
        v = prev_y[(size_t)b * Edim + i];
    } else {
        const int h = i - Edim;
        v = plan[b * 2] * ctx_e[(size_t)b * Hdim + h]
          + plan[b * 2 + 1] * ctx_s[(size_t)b * Hdim + h];
    }
    y_ctx[(size_t)b * (Edim + Hdim) + i] = v;
}

// ---------------------- GRU gate elementwise -------------------------------
__global__ void gru_gate(const float* __restrict__ gx, const float* __restrict__ gh,
                         const float* __restrict__ prev_h,
                         float* __restrict__ h_ws, float* __restrict__ h_out) {
    const size_t idx = (size_t)blockIdx.x * blockDim.x + threadIdx.x;  // BS*D
    const int b = (int)(idx / Ddim), d = (int)(idx % Ddim);
    const size_t base = (size_t)b * 3 * Ddim;
    const float r = sigmoidf(gx[base + d] + gh[base + d]);
    const float z = sigmoidf(gx[base + Ddim + d] + gh[base + Ddim + d]);
    const float n = tanhf(gx[base + 2 * Ddim + d] + r * gh[base + 2 * Ddim + d]);
    const float out = (1.0f - z) * n + z * prev_h[idx];
    h_ws[idx] = out;
    h_out[idx] = out;
}

// ---------------------------------------------------------------------------
extern "C" void kernel_launch(void* const* d_in, const int* in_sizes, int n_in,
                              void* d_out, int out_size, void* d_ws, size_t ws_size,
                              hipStream_t stream) {
    const float* prev_y  = (const float*)d_in[0];
    const float* prev_h  = (const float*)d_in[1];
    const float* equ_enc = (const float*)d_in[2];
    const float* sns_enc = (const float*)d_in[3];
    // d_in[4] = z_sample (unused by reference)
    const int*   equ_mask = (const int*)d_in[5];
    const int*   sns_mask = (const int*)d_in[6];
    const float* W1e = (const float*)d_in[7];
    const float* W2e = (const float*)d_in[8];
    const float* vte = (const float*)d_in[9];
    const float* W1s = (const float*)d_in[10];
    const float* W2s = (const float*)d_in[11];
    const float* vts = (const float*)d_in[12];
    const float* plan_W = (const float*)d_in[13];
    const float* plan_b = (const float*)d_in[14];
    const float* Wc  = (const float*)d_in[15];
    const float* bc  = (const float*)d_in[16];
    const float* w_ih = (const float*)d_in[17];
    const float* w_hh = (const float*)d_in[18];
    const float* b_ih = (const float*)d_in[19];
    const float* b_hh = (const float*)d_in[20];
    const float* Wout = (const float*)d_in[21];
    const float* bout = (const float*)d_in[22];

    float* out = (float*)d_out;
    float* dec_output = out;                                  // (BS, V)
    float* dec_hidden = out + (size_t)BS * Vdim;              // (BS, D)
    float* plan_out   = dec_hidden + (size_t)BS * Ddim;       // (BS, 2)

    // workspace layout (floats)
    float* ws   = (float*)d_ws;
    float* dt_e  = ws;                       // 256*512
    float* dt_s  = dt_e  + (size_t)BS * Hdim;
    float* u_e   = dt_s  + (size_t)BS * Hdim;      // 256*128
    float* u_s   = u_e   + (size_t)BS * Ndim;
    float* aw_e  = u_s   + (size_t)BS * Ndim;
    float* aw_s  = aw_e  + (size_t)BS * Ndim;
    float* ctx_e = aw_s  + (size_t)BS * Ndim;      // 256*512
    float* ctx_s = ctx_e + (size_t)BS * Hdim;
    float* plan  = ctx_s + (size_t)BS * Hdim;      // 256*2
    float* y_ctx = plan  + (size_t)BS * 2;         // 256*1024
    float* xbuf  = y_ctx + (size_t)BS * (Edim + Hdim);  // 256*512
    float* gx    = xbuf  + (size_t)BS * Hdim;      // 256*3456
    float* gh    = gx    + (size_t)BS * 3 * Ddim;  // 256*3456
    float* h_new = gh    + (size_t)BS * 3 * Ddim;  // 256*1152

    const dim3 blk128(128), blk256(256);

    // dt = prev_h @ W2^T  (M=256, N=512, K=1152)
    gemm_bt_bias<<<dim3(BS / 16, Hdim / 64), blk128, 0, stream>>>(prev_h, W2e, nullptr, dt_e, BS, Hdim, Ddim);
    gemm_bt_bias<<<dim3(BS / 16, Hdim / 64), blk128, 0, stream>>>(prev_h, W2s, nullptr, dt_s, BS, Hdim, Ddim);

    // fused attention scores (32768 rows, 512 cols, K=512)
    attn_scores<<<dim3(BS * Ndim / 16), blk128, 0, stream>>>(equ_enc, W1e, dt_e, vte, u_e);
    attn_scores<<<dim3(BS * Ndim / 16), blk128, 0, stream>>>(sns_enc, W1s, dt_s, vts, u_s);

    masked_softmax<<<dim3(BS), blk128, 0, stream>>>(u_e, equ_mask, aw_e);
    masked_softmax<<<dim3(BS), blk128, 0, stream>>>(u_s, sns_mask, aw_s);

    weighted_ctx<<<dim3(BS, Hdim / 256), blk256, 0, stream>>>(aw_e, equ_enc, ctx_e);
    weighted_ctx<<<dim3(BS, Hdim / 256), blk256, 0, stream>>>(aw_s, sns_enc, ctx_s);

    plan_kernel<<<dim3(BS), blk128, 0, stream>>>(prev_h, plan_W, plan_b, plan, plan_out);

    build_yctx<<<dim3(BS, (Edim + Hdim) / 256), blk256, 0, stream>>>(prev_y, plan, ctx_e, ctx_s, y_ctx);

    // x = y_ctx @ Wc^T + bc   (M=256, N=512, K=1024)
    gemm_bt_bias<<<dim3(BS / 16, Hdim / 64), blk128, 0, stream>>>(y_ctx, Wc, bc, xbuf, BS, Hdim, Edim + Hdim);

    // gx = x @ w_ih^T + b_ih  (N=3456, K=512); gh = prev_h @ w_hh^T + b_hh (K=1152)
    gemm_bt_bias<<<dim3(BS / 16, (3 * Ddim) / 64), blk128, 0, stream>>>(xbuf, w_ih, b_ih, gx, BS, 3 * Ddim, Hdim);
    gemm_bt_bias<<<dim3(BS / 16, (3 * Ddim) / 64), blk128, 0, stream>>>(prev_h, w_hh, b_hh, gh, BS, 3 * Ddim, Ddim);

    gru_gate<<<dim3((BS * Ddim) / 256), blk256, 0, stream>>>(gx, gh, prev_h, h_new, dec_hidden);

    // dec_output = h_new @ Wout^T + bout  (M=256, N=32000, K=1152)
    gemm_bt_bias<<<dim3(BS / 16, Vdim / 64), blk128, 0, stream>>>(h_new, Wout, bout, dec_output, BS, Vdim, Ddim);
}
